// MultiHeadedRelativeAttention_56470230008598
// MI455X (gfx1250) — matrix operand order repaired and checked by reading, not compiled, not run on hardware
//
#include <hip/hip_runtime.h>
#include <hip/hip_bf16.h>

typedef __attribute__((ext_vector_type(16))) _Float16 v16h;
typedef __attribute__((ext_vector_type(8)))  _Float16 v8h;
typedef __attribute__((ext_vector_type(8)))  float    v8f;
typedef __attribute__((ext_vector_type(4)))  int      v4i;

#define WMMA_F16(A, Bv, C) \
  __builtin_amdgcn_wmma_f32_16x16x32_f16(false, (A), false, (Bv), (short)0, (C), false, false)

#define CAT16(lo, hi) \
  __builtin_shufflevector((lo), (hi), 0,1,2,3,4,5,6,7,8,9,10,11,12,13,14,15)

namespace {
constexpr int kB  = 2;
constexpr int kH  = 16;
constexpr int kL  = 1024;
constexpr int kD  = 1024;
constexpr int kDK = 64;
constexpr int kRP = 160;   // relative-index dim 129 padded to multiple of 32
}

__device__ __forceinline__ int imin(int a, int b) { return a < b ? a : b; }

// Faithful to reference _toeplitz: aK[l,j] = wK[rel_idx(l,j)]
__device__ __forceinline__ int rel_idx(int l, int j) {
  int d = j - l;
  if (d >= 0) return 64 + imin(d, 64);        // 64..128
  return imin(l - j, 64) - 1;                 // 0..63
}

// ---- CDNA5 async global->LDS copy (ASYNCcnt path), with safe fallback ----

__device__ __forceinline__ void cp_async_b128(const void* g, void* l) {
#if __has_builtin(__builtin_amdgcn_global_load_async_to_lds_b128)
  __builtin_amdgcn_global_load_async_to_lds_b128((v4i*)g, (v4i*)l, 0, 0);
#else
  *(v8h*)l = *(const v8h*)g;
#endif
}

__device__ __forceinline__ void wait_async0() {
#if __has_builtin(__builtin_amdgcn_s_wait_asynccnt)
  __builtin_amdgcn_s_wait_asynccnt(0);
#else
  asm volatile("" ::: "memory");
#endif
}

// ---------------- conversion kernels ----------------

__global__ void cvt_f16_kernel(const float* __restrict__ in, _Float16* __restrict__ out, int n) {
  for (int i = blockIdx.x * blockDim.x + threadIdx.x; i < n; i += gridDim.x * blockDim.x)
    out[i] = (_Float16)in[i];
}

// WT[k*n + c] = W[c*n + k]  (so WT is the K x N operand of  x @ W^T)
__global__ void transpose_f16_kernel(const float* __restrict__ W, _Float16* __restrict__ WT, int n) {
  for (int i = blockIdx.x * blockDim.x + threadIdx.x; i < n * n; i += gridDim.x * blockDim.x) {
    int k = i / n, c = i % n;
    WT[k * n + c] = (_Float16)W[c * n + k];
  }
}

// wk16[r][d] (r-major, zero padded r>=129); wvt16[d][r] (transposed, padded)
__global__ void pack_w_kernel(const float* __restrict__ wK, const float* __restrict__ wV,
                              _Float16* __restrict__ wk16, _Float16* __restrict__ wvt16) {
  for (int i = blockIdx.x * blockDim.x + threadIdx.x; i < kRP * kDK; i += gridDim.x * blockDim.x) {
    int r = i / kDK, d = i % kDK;
    float a = (r < 2 * 64 + 1) ? wK[r * kDK + d] : 0.0f;
    float c = (r < 2 * 64 + 1) ? wV[r * kDK + d] : 0.0f;
    wk16[r * kDK + d]  = (_Float16)a;
    wvt16[d * kRP + r] = (_Float16)c;
  }
}

// ---------------- WMMA GEMM: out = A(MxK) @ Bm(KxN) + bias ----------------
// mode 0: store f16 in per-head layout [(b*H+h)*L + l]*DK + dk   (n = h*64+dk, m = b*L+l)
// mode 1: store f32 row-major [m*N + n]

__global__ void __launch_bounds__(256)
gemm16_kernel(const _Float16* __restrict__ A, const _Float16* __restrict__ Bm,
              const float* __restrict__ bias, _Float16* __restrict__ out16,
              float* __restrict__ out32, int M, int N, int Kdim, int mode) {
  __shared__ _Float16 sa[64 * 32];   // [m][k]
  __shared__ _Float16 sb[64 * 32];   // [n][k] (transposed so B-frag reads are contiguous)
  const int tid  = threadIdx.x;
  const int lane = tid & 31;
  const int wave = tid >> 5;
  const int mblk = blockIdx.y * 64;
  const int nblk = blockIdx.x * 64;
  const int mrow = (wave >> 1) * 16;
  const int n0   = (wave & 1) * 32;
  const int mloc = lane & 15;
  const int aoff = (lane < 16) ? 0 : 16;
  const int boff = (lane < 16) ? 0 : 16;
  v8f acc0 = {}, acc1 = {};
  for (int k0 = 0; k0 < Kdim; k0 += 32) {
    __syncthreads();
    {
      // A tile: contiguous 16B per thread -> async global->LDS (no VGPR round trip)
      int row = tid >> 2, c = (tid & 3) * 8;
      cp_async_b128(&A[(size_t)(mblk + row) * Kdim + k0 + c], &sa[row * 32 + c]);
    }
    {
      // B tile needs a transpose -> classic VGPR staging with ds scatter
      int k = tid >> 3, c = (tid & 7) * 8;
      v8h t = *(const v8h*)&Bm[(size_t)(k0 + k) * N + nblk + c];
#pragma unroll
      for (int u = 0; u < 8; ++u) sb[(c + u) * 32 + k] = t[u];
    }
    wait_async0();
    __syncthreads();
    v8h alo = *(const v8h*)&sa[(mrow + mloc) * 32 + aoff];
    v8h ahi = *(const v8h*)&sa[(mrow + mloc) * 32 + 8 + aoff];
    v16h af = CAT16(alo, ahi);
    v16h b0 = *(const v16h*)&sb[(n0 + mloc) * 32 + boff];
    v16h b1 = *(const v16h*)&sb[(n0 + 16 + mloc) * 32 + boff];
    acc0 = WMMA_F16(af, b0, acc0);
    acc1 = WMMA_F16(af, b1, acc1);
  }
  const int mo = (lane < 16) ? 0 : 8;
#pragma unroll
  for (int i = 0; i < 8; ++i) {
    int m   = mblk + mrow + mo + i;
    int n_a = nblk + n0 + mloc;
    int n_b = n_a + 16;
    float va = acc0[i] + bias[n_a];
    float vb = acc1[i] + bias[n_b];
    if (mode == 0) {
      int bI = m >> 10, l = m & 1023;
      int ha = n_a >> 6, da = n_a & 63;
      int hb = n_b >> 6, db = n_b & 63;
      out16[(((size_t)bI * kH + ha) * kL + l) * kDK + da] = (_Float16)va;
      out16[(((size_t)bI * kH + hb) * kL + l) * kDK + db] = (_Float16)vb;
    } else {
      out32[(size_t)m * N + n_a] = va;
      out32[(size_t)m * N + n_b] = vb;
    }
  }
}

// ---------------- attention kernel ----------------
// grid: B*H*(L/64), block: 128 (4 waves); each wave owns 16 query rows.

__global__ void __launch_bounds__(128)
attn_kernel(const _Float16* __restrict__ qh, const _Float16* __restrict__ kh,
            const _Float16* __restrict__ vh, const int* __restrict__ mask,
            const _Float16* __restrict__ wk16, const _Float16* __restrict__ wvt16,
            _Float16* __restrict__ xh) {
  extern __shared__ char smem_raw[];
  _Float16* s_wk  = (_Float16*)smem_raw;              // [kRP][64]
  _Float16* s_wvt = s_wk + kRP * kDK;                 // [64][kRP]
  _Float16* s_kt  = s_wvt + kDK * kRP;                // [64 j][64 d]
  _Float16* s_vt  = s_kt + 64 * 64;                   // [64 d][64 j] (transposed)
  _Float16* s_S   = s_vt + 64 * 64;                   // [4 waves][16 rows][kL]
  float*    s_qw  = (float*)(s_S + 4 * 16 * kL);      // [4][16][kRP]
  float*    s_pw  = s_qw + 4 * 16 * kRP;              // [4][16][kRP]

  const int tid  = threadIdx.x;
  const int wave = tid >> 5;
  const int lane = tid & 31;
  const int bh   = blockIdx.x >> 4;          // L/64 == 16 row-blocks per (b,h)
  const int lblk = (blockIdx.x & 15) * 64;
  const int b    = bh >> 4;                  // H == 16
  const int h    = bh & 15;
  const int l0   = lblk + wave * 16;
  const int mloc = lane & 15;
  const int aoff = (lane < 16) ? 0 : 16;
  const int boff = (lane < 16) ? 0 : 16;
  const int mo   = (lane < 16) ? 0 : 8;

  // stage relative-weight tables (contiguous -> async copies)
  for (int i = tid; i < (kRP * kDK) / 8; i += 128) {
    cp_async_b128(&wk16[i * 8],  &s_wk[i * 8]);
    cp_async_b128(&wvt16[i * 8], &s_wvt[i * 8]);
  }
  // zero own pw region
  for (int i = lane; i < 16 * kRP; i += 32) s_pw[wave * 16 * kRP + i] = 0.0f;

  // load Q A-fragments (DK=64 -> two K=32 chunks), contiguous k = 16*half + s order (same as B)
  const _Float16* qrow = qh + ((size_t)bh * kL + l0 + mloc) * kDK;
  v16h qa0 = CAT16(*(const v8h*)&qrow[aoff],      *(const v8h*)&qrow[8 + aoff]);
  v16h qa1 = CAT16(*(const v8h*)&qrow[32 + aoff], *(const v8h*)&qrow[40 + aoff]);

  wait_async0();
  __syncthreads();

  // qw = Q @ wK^T  (N = kRP, padded)
  for (int rt = 0; rt < kRP / 16; ++rt) {
    int rc = rt * 16 + mloc;
    v16h b0 = *(const v16h*)&s_wk[rc * kDK + boff];
    v16h b1 = *(const v16h*)&s_wk[rc * kDK + 32 + boff];
    v8f acc = {};
    acc = WMMA_F16(qa0, b0, acc);
    acc = WMMA_F16(qa1, b1, acc);
#pragma unroll
    for (int i = 0; i < 8; ++i)
      s_qw[(wave * 16 + mo + i) * kRP + rt * 16 + mloc] = acc[i];
  }

  // scores: S = (Q K^T + rprK) / sqrt(DK), masked, stored f16 in LDS
  for (int jb = 0; jb < kL; jb += 64) {
    __syncthreads();
    for (int i = tid; i < 512; i += 128) {
      int j = i >> 3, c = (i & 7) * 8;
      cp_async_b128(&kh[((size_t)bh * kL + jb + j) * kDK + c], &s_kt[j * 64 + c]);
    }
    wait_async0();
    __syncthreads();
    for (int nt = 0; nt < 4; ++nt) {
      int jl = nt * 16 + mloc;
      v16h b0 = *(const v16h*)&s_kt[jl * 64 + boff];
      v16h b1 = *(const v16h*)&s_kt[jl * 64 + 32 + boff];
      v8f acc = {};
      acc = WMMA_F16(qa0, b0, acc);
      acc = WMMA_F16(qa1, b1, acc);
      int jg = jb + nt * 16 + mloc;
#pragma unroll
      for (int i = 0; i < 8; ++i) {
        int l = l0 + mo + i;
        float rpr = s_qw[(wave * 16 + mo + i) * kRP + rel_idx(l, jg)];
        float sv = (acc[i] + rpr) * 0.125f;
        if (mask[(b * kL + l) * kL + jg] == 0) sv = -1.0e9f;
        s_S[(wave * 16 + mo + i) * kL + jg] = (_Float16)sv;
      }
    }
  }

  // softmax per row (2 lanes per row) + Toeplitz binning into pw
  {
    int row  = lane >> 1;
    int half = lane & 1;
    int l    = l0 + row;
    _Float16* Srow = s_S + (wave * 16 + row) * kL;
    float* pwrow   = s_pw + (wave * 16 + row) * kRP;
    int jstart = half * 512;
    float mx = -3.0e38f;
    for (int j = jstart; j < jstart + 512; ++j) mx = fmaxf(mx, (float)Srow[j]);
    mx = fmaxf(mx, __shfl_xor(mx, 1, 32));
    float sum = 0.0f;
    for (int j = jstart; j < jstart + 512; ++j) sum += __expf((float)Srow[j] - mx);
    sum += __shfl_xor(sum, 1, 32);
    float inv = 1.0f / sum;
    float farL = 0.0f, farR = 0.0f;
    for (int j = jstart; j < jstart + 512; ++j) {
      float p = __expf((float)Srow[j] - mx) * inv;
      Srow[j] = (_Float16)p;
      int r = rel_idx(l, j);
      if (r == 63)       farL += p;        // all j <= l-64
      else if (r == 128) farR += p;        // all j >= l+64
      else               pwrow[r] += p;    // unique j per bin -> single writer
    }
    farL += __shfl_xor(farL, 1, 32);
    farR += __shfl_xor(farR, 1, 32);
    if (half == 0) { pwrow[63] += farL; pwrow[128] += farR; }
  }

  // O = P @ V (accumulated over j), then O += pw @ wV
  v8f o0 = {}, o1 = {}, o2 = {}, o3 = {};
  for (int jb = 0; jb < kL; jb += 64) {
    __syncthreads();
    for (int i = tid; i < 512; i += 128) {
      int j = i >> 3, c = (i & 7) * 8;
      v8h t = *(const v8h*)&vh[((size_t)bh * kL + jb + j) * kDK + c];
#pragma unroll
      for (int u = 0; u < 8; ++u) s_vt[(c + u) * 64 + j] = t[u];   // transpose scatter
    }
    __syncthreads();
    for (int jc = 0; jc < 2; ++jc) {
      const _Float16* srow = s_S + (wave * 16 + mloc) * kL + jb + jc * 32;
      v16h pa = CAT16(*(const v8h*)&srow[aoff], *(const v8h*)&srow[8 + aoff]);
      v16h vb;
      vb = *(const v16h*)&s_vt[(0  + mloc) * 64 + jc * 32 + boff]; o0 = WMMA_F16(pa, vb, o0);
      vb = *(const v16h*)&s_vt[(16 + mloc) * 64 + jc * 32 + boff]; o1 = WMMA_F16(pa, vb, o1);
      vb = *(const v16h*)&s_vt[(32 + mloc) * 64 + jc * 32 + boff]; o2 = WMMA_F16(pa, vb, o2);
      vb = *(const v16h*)&s_vt[(48 + mloc) * 64 + jc * 32 + boff]; o3 = WMMA_F16(pa, vb, o3);
    }
  }
  for (int rc2 = 0; rc2 < kRP / 32; ++rc2) {
    const float* prow = s_pw + (wave * 16 + mloc) * kRP + rc2 * 32;
    v16h pa;
#pragma unroll
    for (int i = 0; i < 8; ++i) {
      pa[i]     = (_Float16)prow[aoff + i];
      pa[8 + i] = (_Float16)prow[8 + aoff + i];
    }
    v16h vb;
    vb = *(const v16h*)&s_wvt[(0  + mloc) * kRP + rc2 * 32 + boff]; o0 = WMMA_F16(pa, vb, o0);
    vb = *(const v16h*)&s_wvt[(16 + mloc) * kRP + rc2 * 32 + boff]; o1 = WMMA_F16(pa, vb, o1);
    vb = *(const v16h*)&s_wvt[(32 + mloc) * kRP + rc2 * 32 + boff]; o2 = WMMA_F16(pa, vb, o2);
    vb = *(const v16h*)&s_wvt[(48 + mloc) * kRP + rc2 * 32 + boff]; o3 = WMMA_F16(pa, vb, o3);
  }

#pragma unroll
  for (int i = 0; i < 8; ++i) {
    int l = l0 + mo + i;
    size_t base = ((size_t)b * kL + l) * kD + h * kDK + mloc;
    xh[base + 0]  = (_Float16)o0[i];
    xh[base + 16] = (_Float16)o1[i];
    xh[base + 32] = (_Float16)o2[i];
    xh[base + 48] = (_Float16)o3[i];
  }
}

// ---------------- launcher ----------------

extern "C" void kernel_launch(void* const* d_in, const int* in_sizes, int n_in,
                              void* d_out, int out_size, void* d_ws, size_t ws_size,
                              hipStream_t stream) {
  (void)in_sizes; (void)n_in; (void)out_size; (void)ws_size;
  const float* query = (const float*)d_in[0];
  const float* key_  = (const float*)d_in[1];
  const float* value = (const float*)d_in[2];
  const int*   mask  = (const int*)d_in[3];
  const float* Wq = (const float*)d_in[4];  const float* bq = (const float*)d_in[5];
  const float* Wk = (const float*)d_in[6];  const float* bk = (const float*)d_in[7];
  const float* Wv = (const float*)d_in[8];  const float* bv = (const float*)d_in[9];
  const float* Wo = (const float*)d_in[10]; const float* bo = (const float*)d_in[11];
  const float* wKp = (const float*)d_in[12];
  const float* wVp = (const float*)d_in[13];

  char* ws = (char*)d_ws;
  size_t off = 0;
  auto take = [&](size_t bytes) -> void* {
    void* p = (void*)(ws + off);
    off += (bytes + 255) & ~(size_t)255;
    return p;
  };
  const size_t nBLD = (size_t)kB * kL * kD;          // 2,097,152
  _Float16* q16  = (_Float16*)take(nBLD * 2);
  _Float16* k16  = (_Float16*)take(nBLD * 2);
  _Float16* v16  = (_Float16*)take(nBLD * 2);
  _Float16* WqT  = (_Float16*)take((size_t)kD * kD * 2);
  _Float16* WkT  = (_Float16*)take((size_t)kD * kD * 2);
  _Float16* WvT  = (_Float16*)take((size_t)kD * kD * 2);
  _Float16* WoT  = (_Float16*)take((size_t)kD * kD * 2);
  _Float16* qhd  = (_Float16*)take(nBLD * 2);
  _Float16* khd  = (_Float16*)take(nBLD * 2);
  _Float16* vhd  = (_Float16*)take(nBLD * 2);
  _Float16* xh   = (_Float16*)take(nBLD * 2);
  _Float16* wk16  = (_Float16*)take((size_t)kRP * kDK * 2);
  _Float16* wvt16 = (_Float16*)take((size_t)kRP * kDK * 2);

  cvt_f16_kernel<<<1024, 256, 0, stream>>>(query, q16, (int)nBLD);
  cvt_f16_kernel<<<1024, 256, 0, stream>>>(key_,  k16, (int)nBLD);
  cvt_f16_kernel<<<1024, 256, 0, stream>>>(value, v16, (int)nBLD);
  transpose_f16_kernel<<<1024, 256, 0, stream>>>(Wq, WqT, kD);
  transpose_f16_kernel<<<1024, 256, 0, stream>>>(Wk, WkT, kD);
  transpose_f16_kernel<<<1024, 256, 0, stream>>>(Wv, WvT, kD);
  transpose_f16_kernel<<<1024, 256, 0, stream>>>(Wo, WoT, kD);
  pack_w_kernel<<<40, 256, 0, stream>>>(wKp, wVp, wk16, wvt16);

  dim3 ggrid(kD / 64, (kB * kL) / 64);   // (16, 32)
  gemm16_kernel<<<ggrid, 256, 0, stream>>>(q16, WqT, bq, qhd, nullptr, kB * kL, kD, kD, 0);
  gemm16_kernel<<<ggrid, 256, 0, stream>>>(k16, WkT, bk, khd, nullptr, kB * kL, kD, kD, 0);
  gemm16_kernel<<<ggrid, 256, 0, stream>>>(v16, WvT, bv, vhd, nullptr, kB * kL, kD, kD, 0);

  constexpr int SMEM_BYTES =
      (kRP * kDK * 2) * 2 +        // s_wk + s_wvt
      (64 * 64 * 2) * 2 +          // s_kt + s_vt
      4 * 16 * kL * 2 +            // s_S
      4 * 16 * kRP * 4 * 2;        // s_qw + s_pw     => 270,336 bytes (< 320 KB)
  (void)hipFuncSetAttribute((const void*)attn_kernel,
                            hipFuncAttributeMaxDynamicSharedMemorySize, SMEM_BYTES);
  attn_kernel<<<kB * kH * (kL / 64), 128, SMEM_BYTES, stream>>>(
      qhd, khd, vhd, mask, wk16, wvt16, xh);

  gemm16_kernel<<<ggrid, 256, 0, stream>>>(xh, WoT, bo, nullptr, (float*)d_out,
                                           kB * kL, kD, kD, 1);
}
